// GRAND_24378234372225
// MI455X (gfx1250) — matrix operand order repaired and checked by reading, not compiled, not run on hardware
//
#include <hip/hip_runtime.h>
#include <hip/hip_bf16.h>
#include <math.h>

// MI455X / gfx1250, wave32. Matrix path: v_wmma_f32_16x16x32_bf16.
// One workgroup (8 waves) per sample; wave w handles attention head h=w.
// Weights are pre-converted to bf16 once per launch (d_ws), so hot loops do
// no f32->bf16 conversion on the B operands.

typedef __attribute__((ext_vector_type(16))) __bf16 v16bf;
typedef __attribute__((ext_vector_type(8)))  float  v8f;

#define SMEM_BYTES (256 * 1024)

// Hardware round-to-nearest-even f32 -> bf16 (native cvt on gfx1250).
__device__ __forceinline__ unsigned short f2bf(float f) {
  union { __bf16 b; unsigned short u; } c;
  c.b = (__bf16)f;
  return c.u;
}

union FragBF { v16bf v; uint4 q[2]; unsigned short u[16]; };

// A fragment (16x32 bf16, M x K): lane holds row m = (lane&15); half = lane>>4.
// Elements 0..7  = K [kbase + half*8      .. +7]
// Elements 8..15 = K [kbase + 16 + half*8 .. +7]   (per ISA 16-bit A layout)
__device__ __forceinline__ v16bf load_a(const unsigned short* rowptr, int kbase, int half) {
  FragBF r;
  r.q[0] = *(const uint4*)(rowptr + kbase + half * 8);
  r.q[1] = *(const uint4*)(rowptr + kbase + 16 + half * 8);
  return r.v;
}

// B fragment (32x16 bf16, K x N): lane holds column n = (lane&15); khalf = lane>>4.
// Elements 0..7 = K [kbase + khalf*8 .. +7], 8..15 = K [kbase + 16 + khalf*8 .. +7]
// (same order as A). Generic pointer: works for LDS and global (bf16 weights).
__device__ __forceinline__ v16bf load_b(const unsigned short* colptr, int kbase, int khalf) {
  FragBF r;
  const unsigned short* p = colptr + kbase + khalf * 8;
  r.q[0] = *(const uint4*)(p);
  r.q[1] = *(const uint4*)(p + 16);
  return r.v;
}

__device__ __forceinline__ v8f wmma_bf16(v16bf a, v16bf b, v8f c) {
  return __builtin_amdgcn_wmma_f32_16x16x32_bf16(false, a, false, b, (short)0, c, false, false);
}

__global__ __launch_bounds__(256) void ode_block_kernel(
    const float* __restrict__ X,                         // [N,64,64] f32
    const unsigned short* __restrict__ Wq, const float* __restrict__ bq,   // W* are bf16
    const unsigned short* __restrict__ Wk, const float* __restrict__ bk,
    const unsigned short* __restrict__ Wv, const float* __restrict__ bv,
    const unsigned short* __restrict__ W0, const float* __restrict__ b0,
    float* __restrict__ Y)                               // [N,64,64] f32
{
  extern __shared__ unsigned short smem[];
  unsigned short* regA = smem;            // phase1: xbf (4096); phase2+: att (8*4096)
  unsigned short* qlin = smem + 32768;    // [64][512] bf16
  unsigned short* klin = smem + 65536;    // [64][512] bf16 ; phase3+: vals2 alias
  unsigned short* vT   = smem + 98304;    // [8][64][64] bf16, per-head transposed V

  const int tid  = threadIdx.x;
  const int wave = tid >> 5;
  const int lane = tid & 31;
  const int nl   = lane & 15;
  const int half = lane >> 4;
  const float* xg = X + (size_t)blockIdx.x * 4096;

  // ---- Phase 0: x -> bf16 LDS ----
  for (int i = tid; i < 4096; i += 256) regA[i] = f2bf(xg[i]);
  __syncthreads();

  // ---- Phase 1: qlin/klin (row-major) and vT (per-head transposed) ----
  // GEMM: [64 x 64] @ [64 x 512] per projection; tiles 4 x 32; K-loop of 2.
  // Wave w owns tn columns {w, w+8, w+16, w+24}; A frags hoisted per tm.
  const unsigned short* Wp[3] = {Wq, Wk, Wv};
  const float* bp[3] = {bq, bk, bv};
  for (int wi = 0; wi < 3; ++wi) {
#pragma unroll
    for (int tm = 0; tm < 4; ++tm) {
      const unsigned short* arow = regA + (tm * 16 + nl) * 64;
      v16bf a0 = load_a(arow, 0, half);
      v16bf a1 = load_a(arow, 32, half);
#pragma unroll
      for (int ti = 0; ti < 4; ++ti) {
        const int tn = wave + ti * 8;
        const int nj = tn * 16 + nl;
        const float bias = bp[wi][nj];
        v8f acc = {bias, bias, bias, bias, bias, bias, bias, bias};
        const unsigned short* wrow = Wp[wi] + nj * 64;
        acc = wmma_bf16(a0, load_b(wrow, 0, half), acc);
        acc = wmma_bf16(a1, load_b(wrow, 32, half), acc);
        if (wi < 2) {
          unsigned short* dst = (wi == 0) ? qlin : klin;
#pragma unroll
          for (int p = 0; p < 8; ++p) {
            const int r = tm * 16 + half * 8 + p;
            dst[r * 512 + nj] = f2bf(acc[p]);
          }
        } else {
          // vT[h][f][c2]: h=r>>3, f=nj&63, c2=(r&7)*8 + (nj>>6)
#pragma unroll
          for (int p = 0; p < 8; ++p) {
            const int r = tm * 16 + half * 8 + p;
            vT[(r >> 3) * 4096 + (nj & 63) * 64 + (r & 7) * 8 + (nj >> 6)] = f2bf(acc[p]);
          }
        }
      }
    }
  }
  __syncthreads();

  // ---- Phase 2: per-head attention (wave w == head h) ----
  const int h = wave;
  unsigned short* att = regA + h * 4096;   // [64][64] bf16, head-private

  v8f S[4][4];
  {
    v8f z = {0.f, 0.f, 0.f, 0.f, 0.f, 0.f, 0.f, 0.f};
#pragma unroll
    for (int tm = 0; tm < 4; ++tm)
#pragma unroll
      for (int tn = 0; tn < 4; ++tn) S[tm][tn] = z;
  }
  // Q_h row c lives at qlin[(8h + c/8)*512 + (c%8)*64 + d] (contiguous in d).
#pragma unroll
  for (int tm = 0; tm < 4; ++tm) {
    const int c = tm * 16 + nl;
    const unsigned short* qrow = qlin + (8 * h + (c >> 3)) * 512 + (c & 7) * 64;
#pragma unroll
    for (int kt = 0; kt < 2; ++kt) {
      v16bf a = load_a(qrow, kt * 32, half);
#pragma unroll
      for (int tn = 0; tn < 4; ++tn) {
        const int c2 = tn * 16 + nl;
        const unsigned short* krow = klin + (8 * h + (c2 >> 3)) * 512 + (c2 & 7) * 64;
        v16bf b = load_b(krow, kt * 32, half);
        S[tm][tn] = wmma_bf16(a, b, S[tm][tn]);
      }
    }
  }

  // softmax -> threshold(>0.7) on scaled scores -> softmax, per row, in registers.
  // C-frag row (tm*16 + half*8 + p) is spread across the 16 lanes of this half.
#pragma unroll
  for (int tm = 0; tm < 4; ++tm) {
#pragma unroll
    for (int p = 0; p < 8; ++p) {
      float s[4];
#pragma unroll
      for (int tn = 0; tn < 4; ++tn) s[tn] = S[tm][tn][p] * 0.125f;  // /sqrt(64)
      float mx = fmaxf(fmaxf(s[0], s[1]), fmaxf(s[2], s[3]));
#pragma unroll
      for (int off = 1; off < 16; off <<= 1) mx = fmaxf(mx, __shfl_xor(mx, off, 16));
      float e[4], sum = 0.f;
#pragma unroll
      for (int tn = 0; tn < 4; ++tn) { e[tn] = __expf(s[tn] - mx); sum += e[tn]; }
#pragma unroll
      for (int off = 1; off < 16; off <<= 1) sum += __shfl_xor(sum, off, 16);
      const float inv = 1.f / sum;
      float tt[4];
#pragma unroll
      for (int tn = 0; tn < 4; ++tn) tt[tn] = (e[tn] * inv > 0.7f) ? s[tn] : 0.f;
      float mx2 = fmaxf(fmaxf(tt[0], tt[1]), fmaxf(tt[2], tt[3]));
#pragma unroll
      for (int off = 1; off < 16; off <<= 1) mx2 = fmaxf(mx2, __shfl_xor(mx2, off, 16));
      float e2[4], sum2 = 0.f;
#pragma unroll
      for (int tn = 0; tn < 4; ++tn) { e2[tn] = __expf(tt[tn] - mx2); sum2 += e2[tn]; }
#pragma unroll
      for (int off = 1; off < 16; off <<= 1) sum2 += __shfl_xor(sum2, off, 16);
      const float inv2 = 1.f / sum2;
      const int c = tm * 16 + half * 8 + p;
#pragma unroll
      for (int tn = 0; tn < 4; ++tn) att[c * 64 + tn * 16 + nl] = f2bf(e2[tn] * inv2);
    }
  }
  __syncthreads();   // att visible; all klin reads done -> safe to alias as vals2

  // ---- Phase 3: O_h = att @ V_h ; store as vals2[c, h*64+f] (alias klin) ----
  unsigned short* vals2 = klin;
  {
    v8f O[4][4];
    v8f z = {0.f, 0.f, 0.f, 0.f, 0.f, 0.f, 0.f, 0.f};
#pragma unroll
    for (int tm = 0; tm < 4; ++tm)
#pragma unroll
      for (int tn = 0; tn < 4; ++tn) O[tm][tn] = z;
    const unsigned short* vTh = vT + h * 4096;
#pragma unroll
    for (int tm = 0; tm < 4; ++tm) {
      const unsigned short* arow = att + (tm * 16 + nl) * 64;
#pragma unroll
      for (int kt = 0; kt < 2; ++kt) {
        v16bf a = load_a(arow, kt * 32, half);
#pragma unroll
        for (int tn = 0; tn < 4; ++tn) {
          v16bf b = load_b(vTh + (tn * 16 + nl) * 64, kt * 32, half);
          O[tm][tn] = wmma_bf16(a, b, O[tm][tn]);
        }
      }
    }
#pragma unroll
    for (int tm = 0; tm < 4; ++tm)
#pragma unroll
      for (int tn = 0; tn < 4; ++tn)
#pragma unroll
        for (int p = 0; p < 8; ++p) {
          const int c = tm * 16 + half * 8 + p;
          vals2[c * 512 + h * 64 + tn * 16 + nl] = f2bf(O[tm][tn][p]);
        }
  }
  __syncthreads();

  // ---- Phase 4: out = vals2 @ W0^T + b0 ; [64 x 512] @ [512 x 64] ----
  float* yb = Y + (size_t)blockIdx.x * 4096;
  for (int t = wave; t < 16; t += 8) {
    const int tm = t >> 2, tn = t & 3;
    const int ncol = tn * 16 + nl;
    const float bias = b0[ncol];
    v8f acc = {bias, bias, bias, bias, bias, bias, bias, bias};
    const unsigned short* arow = vals2 + (tm * 16 + nl) * 512;
    const unsigned short* wrow = W0 + ncol * 512;
#pragma unroll
    for (int kt = 0; kt < 16; ++kt) {
      v16bf a = load_a(arow, kt * 32, half);
      v16bf b = load_b(wrow, kt * 32, half);
      acc = wmma_bf16(a, b, acc);
    }
#pragma unroll
    for (int p = 0; p < 8; ++p)
      yb[(tm * 16 + half * 8 + p) * 64 + ncol] = acc[p];
  }
}

// ---- One-time f32 -> bf16 weight conversion ----
__global__ void cvt_bf16_kernel(const float* __restrict__ src,
                                unsigned short* __restrict__ dst, int n) {
  int i = blockIdx.x * blockDim.x + threadIdx.x;
  if (i < n) dst[i] = f2bf(src[i]);
}

// ---- RK4 combine kernels (dt read from device t[] for graph-capture safety) ----
__global__ void axpy_set_kernel(float* __restrict__ dst, const float* __restrict__ x0,
                                const float* __restrict__ k, const float* __restrict__ t,
                                float coef, int n) {
  int i = blockIdx.x * blockDim.x + threadIdx.x;
  if (i < n) { float dt = t[1] - t[0]; dst[i] = x0[i] + coef * dt * k[i]; }
}

__global__ void axpy_add_kernel(float* __restrict__ dst, const float* __restrict__ k,
                                const float* __restrict__ t, float coef, int n) {
  int i = blockIdx.x * blockDim.x + threadIdx.x;
  if (i < n) { float dt = t[1] - t[0]; dst[i] += coef * dt * k[i]; }
}

extern "C" void kernel_launch(void* const* d_in, const int* in_sizes, int n_in,
                              void* d_out, int out_size, void* d_ws, size_t ws_size,
                              hipStream_t stream) {
  const float* x = (const float*)d_in[0];
  const float* t = (const float*)d_in[1];
  // inputs: x, t, then (enc|dec) x (q_W,q_b,k_W,k_b,v_W,v_b,w0_W,w0_b)
  const float* encf[8]; const float* decf[8];
  for (int i = 0; i < 8; ++i) encf[i] = (const float*)d_in[2 + i];
  for (int i = 0; i < 8; ++i) decf[i] = (const float*)d_in[10 + i];

  const int Ntot = in_sizes[0];            // N*C*F
  const int N = Ntot / 4096;               // samples
  float* z    = (float*)d_out;             // first output
  float* xhat = z + Ntot;                  // second output
  float* kbuf = (float*)d_ws;
  float* tmp  = kbuf + Ntot;
  unsigned short* wbf = (unsigned short*)(tmp + Ntot);   // 8 x 32768 bf16

  (void)hipFuncSetAttribute(reinterpret_cast<const void*>(ode_block_kernel),
                            hipFuncAttributeMaxDynamicSharedMemorySize, SMEM_BYTES);

  // Convert the 8 weight matrices (each 512x64 or 64x512 = 32768 elems) to bf16.
  const int WN = 32768;
  const float* wsrc[8] = {encf[0], encf[2], encf[4], encf[6],
                          decf[0], decf[2], decf[4], decf[6]};
  for (int i = 0; i < 8; ++i)
    cvt_bf16_kernel<<<(WN + 255) / 256, 256, 0, stream>>>(wsrc[i], wbf + (size_t)i * WN, WN);

  struct BW { const unsigned short* W[4]; const float* b[4]; };
  BW enc = {{wbf + 0 * WN, wbf + 1 * WN, wbf + 2 * WN, wbf + 3 * WN},
            {encf[1], encf[3], encf[5], encf[7]}};
  BW dec = {{wbf + 4 * WN, wbf + 5 * WN, wbf + 6 * WN, wbf + 7 * WN},
            {decf[1], decf[3], decf[5], decf[7]}};

  const int EB = 256, EG = (Ntot + EB - 1) / EB;
  const float c16 = 1.f / 6.f, c13 = 1.f / 3.f;

  auto run_block = [&](const float* xin, const BW& w, float* kout) {
    ode_block_kernel<<<N, 256, SMEM_BYTES, stream>>>(
        xin, w.W[0], w.b[0], w.W[1], w.b[1], w.W[2], w.b[2], w.W[3], w.b[3], kout);
  };

  auto run_ode = [&](const float* x0, const BW& w, float* dst) {
    run_block(x0, w, kbuf);                                              // k1
    axpy_set_kernel<<<EG, EB, 0, stream>>>(dst, x0, kbuf, t, c16, Ntot); // dst = x0 + dt/6 k1
    axpy_set_kernel<<<EG, EB, 0, stream>>>(tmp, x0, kbuf, t, 0.5f, Ntot);
    run_block(tmp, w, kbuf);                                             // k2
    axpy_add_kernel<<<EG, EB, 0, stream>>>(dst, kbuf, t, c13, Ntot);
    axpy_set_kernel<<<EG, EB, 0, stream>>>(tmp, x0, kbuf, t, 0.5f, Ntot);
    run_block(tmp, w, kbuf);                                             // k3
    axpy_add_kernel<<<EG, EB, 0, stream>>>(dst, kbuf, t, c13, Ntot);
    axpy_set_kernel<<<EG, EB, 0, stream>>>(tmp, x0, kbuf, t, 1.0f, Ntot);
    run_block(tmp, w, kbuf);                                             // k4
    axpy_add_kernel<<<EG, EB, 0, stream>>>(dst, kbuf, t, c16, Ntot);
  };

  run_ode(x, enc, z);      // z = odeint(enc, x)
  run_ode(z, dec, xhat);   // x_hat = odeint(dec, z)
}
